// SparseGraphLinkModule_12627203850642
// MI455X (gfx1250) — compile-verified, matrix operand order checked by reading, not run on hardware
//
#include <hip/hip_runtime.h>
#include <hip/hip_bf16.h>

// ---------------------------------------------------------------------------
// SparseGraphLinkModule for MI455X (gfx1250, wave32, WMMA).
// bf16-resident GEMM operands, V_WMMA_F32_16X16X32_BF16 with fp32 accum,
// async global->LDS staging (double buffered) on the NT path, factored
// adjacency so the dense 769x769 adj is never materialized.
// ---------------------------------------------------------------------------

#define B_ 64
#define NV_ 256
#define NK_ 512
#define D_ 1024
#define NT_ 769      // NV + 1 + NK
#define KS_ 257      // NV + 1
#define TOPK_ 8

typedef __attribute__((ext_vector_type(8)))  float  v8f;
typedef __attribute__((ext_vector_type(16))) __bf16 v16bf;
typedef unsigned short ushort_t;

static __device__ inline unsigned short f2bf(float f) {
  unsigned u = __float_as_uint(f);
  unsigned r = u + 0x7FFFu + ((u >> 16) & 1u);   // round-to-nearest-even
  return (unsigned short)(r >> 16);
}

// 256-thread block sum reduction (sm must be float[256])
static __device__ inline float block_sum(float v, float* sm) {
  int t = threadIdx.x;
  sm[t] = v; __syncthreads();
  for (int s = 128; s > 0; s >>= 1) {
    if (t < s) sm[t] += sm[t + s];
    __syncthreads();
  }
  float r = sm[0]; __syncthreads();
  return r;
}

// async 16B global -> LDS copy (CDNA5, tracked by ASYNCcnt)
static __device__ inline void async_cp16(unsigned lds_byte, const ushort_t* gp) {
  asm volatile("global_load_async_to_lds_b128 %0, %1, off"
               :: "v"(lds_byte), "v"(gp) : "memory");
}
static __device__ inline void async_wait0() {
  asm volatile("s_wait_asynccnt 0" ::: "memory");
}

// fp32 -> bf16 bulk conversion (4 elems / thread)
__global__ __launch_bounds__(256) void cvt_bf16(
    const float* __restrict__ s, ushort_t* __restrict__ d, size_t n4)
{
  size_t gid = (size_t)blockIdx.x * 256 + threadIdx.x;
  if (gid >= n4) return;
  float4 v = ((const float4*)s)[gid];
  uint2 p;
  p.x = (unsigned)f2bf(v.x) | ((unsigned)f2bf(v.y) << 16);
  p.y = (unsigned)f2bf(v.z) | ((unsigned)f2bf(v.w) << 16);
  ((uint2*)d)[gid] = p;
}

// ---------------------------------------------------------------------------
// NT GEMM: C[M,N] = A[M,K] @ B[N,K]^T ; A,B bf16, C fp32.
// Tile 128x128x32, 8 waves, 4x2 WMMA accumulators per wave.
// Double-buffered LDS fed by global_load_async_to_lds_b128.
// ---------------------------------------------------------------------------
__global__ __launch_bounds__(256) void gemm_bf16_nt(
    const ushort_t* __restrict__ Ag, size_t sA,
    const ushort_t* __restrict__ Bg, size_t sB,
    float* __restrict__ Cg, size_t sC,
    int M, int N, int K, int lda, int ldb, int ldc)
{
  // rows of 32 bf16 padded to 40 (80B) -> 16B-aligned chunks, no bank hotspots
  __shared__ __align__(16) ushort_t lA[2][128 * 40];
  __shared__ __align__(16) ushort_t lB[2][128 * 40];

  const int tid = threadIdx.x;
  const int lane = tid & 31, wid = tid >> 5;
  const int wm = wid >> 2, wn = wid & 3;
  const int laneLo = lane & 15, laneHi = lane >> 4;
  const int bm0 = blockIdx.y * 128, bn0 = blockIdx.x * 128;

  const ushort_t* A  = Ag + (size_t)blockIdx.z * sA;
  const ushort_t* Bp = Bg + (size_t)blockIdx.z * sB;
  float*          C  = Cg + (size_t)blockIdx.z * sC;

  const unsigned baseA[2] = {(unsigned)(uintptr_t)&lA[0][0],
                             (unsigned)(uintptr_t)&lA[1][0]};
  const unsigned baseB[2] = {(unsigned)(uintptr_t)&lB[0][0],
                             (unsigned)(uintptr_t)&lB[1][0]};

  // stage one 128x32 bf16 tile (rows r0..r0+127, clamped) into LDS buffer
  auto stage = [&](unsigned lbase, const ushort_t* g, int ld, int r0,
                   int lim, int k0) {
#pragma unroll
    for (int j = 0; j < 2; ++j) {
      int idx = tid + j * 256;
      int row = idx >> 2, ch = idx & 3;      // 4 x 16B chunks per 64B row
      int gr = r0 + row; if (gr >= lim) gr = lim - 1;   // branchless clamp
      async_cp16(lbase + row * 80 + ch * 16,
                 g + (size_t)gr * ld + k0 + ch * 8);
    }
  };

  v8f acc[4][2] = {};
  int cur = 0;
  stage(baseA[0], A, lda, bm0, M, 0);
  stage(baseB[0], Bp, ldb, bn0, N, 0);
  async_wait0();
  __syncthreads();

  for (int k0 = 0; k0 < K; k0 += 32) {
    int nxt = cur ^ 1;
    if (k0 + 32 < K) {                       // prefetch next K-tile into LDS
      stage(baseA[nxt], A, lda, bm0, M, k0 + 32);
      stage(baseB[nxt], Bp, ldb, bn0, N, k0 + 32);
    }
    // fragments: lane L holds row/col (L%16); element e covers
    // K = (e%8) + 16*(e/8) + 8*(L/16) -> two contiguous 8-elem chunks
    union Frag { v16bf v; uint4 q[2]; };
    Frag fa[4], fb[2];
#pragma unroll
    for (int tm = 0; tm < 4; ++tm) {
      int row = wm * 64 + tm * 16 + laneLo;
      fa[tm].q[0] = *(const uint4*)&lA[cur][row * 40 + 8 * laneHi];
      fa[tm].q[1] = *(const uint4*)&lA[cur][row * 40 + 16 + 8 * laneHi];
    }
#pragma unroll
    for (int tn = 0; tn < 2; ++tn) {
      int col = wn * 32 + tn * 16 + laneLo;
      fb[tn].q[0] = *(const uint4*)&lB[cur][col * 40 + 8 * laneHi];
      fb[tn].q[1] = *(const uint4*)&lB[cur][col * 40 + 16 + 8 * laneHi];
    }
#pragma unroll
    for (int tm = 0; tm < 4; ++tm)
#pragma unroll
      for (int tn = 0; tn < 2; ++tn)
        acc[tm][tn] = __builtin_amdgcn_wmma_f32_16x16x32_bf16(
            false, fa[tm].v, false, fb[tn].v, (short)0, acc[tm][tn],
            false, false);
    async_wait0();
    __syncthreads();
    cur = nxt;
  }

#pragma unroll
  for (int tm = 0; tm < 4; ++tm) {
#pragma unroll
    for (int tn = 0; tn < 2; ++tn) {
      int gcol = bn0 + wn * 32 + tn * 16 + laneLo;
#pragma unroll
      for (int r = 0; r < 8; ++r) {
        int grow = bm0 + wm * 64 + tm * 16 + r + 8 * laneHi;
        if (grow < M && gcol < N)
          C[(size_t)grow * ldc + gcol] = acc[tm][tn][r];
      }
    }
  }
}

// ---------------------------------------------------------------------------
// NN GEMM: C[M,N] = A[M,K] @ B[K,N] ; A,B bf16, C fp32.
// A staged async; B transposed to [n][k] through VGPRs (async can't transpose).
// ---------------------------------------------------------------------------
__global__ __launch_bounds__(256) void gemm_bf16_nn(
    const ushort_t* __restrict__ Ag, size_t sA,
    const ushort_t* __restrict__ Bg, size_t sB,
    float* __restrict__ Cg, size_t sC,
    int M, int N, int K, int lda, int ldb, int ldc)
{
  __shared__ __align__(16) ushort_t lA[128 * 40];
  __shared__ __align__(16) ushort_t lB[128 * 40];

  const int tid = threadIdx.x;
  const int lane = tid & 31, wid = tid >> 5;
  const int wm = wid >> 2, wn = wid & 3;
  const int laneLo = lane & 15, laneHi = lane >> 4;
  const int bm0 = blockIdx.y * 128, bn0 = blockIdx.x * 128;

  const ushort_t* A  = Ag + (size_t)blockIdx.z * sA;
  const ushort_t* Bp = Bg + (size_t)blockIdx.z * sB;
  float*          C  = Cg + (size_t)blockIdx.z * sC;

  const unsigned baseA = (unsigned)(uintptr_t)&lA[0];

  v8f acc[4][2] = {};

  for (int k0 = 0; k0 < K; k0 += 32) {
    // A rows, direct async copy
#pragma unroll
    for (int j = 0; j < 2; ++j) {
      int idx = tid + j * 256;
      int row = idx >> 2, ch = idx & 3;
      int gr = bm0 + row; if (gr >= M) gr = M - 1;
      async_cp16(baseA + row * 80 + ch * 16,
                 A + (size_t)gr * lda + k0 + ch * 8);
    }
    // B [K,N] -> lB[n][k] transpose through VGPRs
#pragma unroll
    for (int j = 0; j < 2; ++j) {
      int idx = tid + j * 256;
      int kk = idx >> 4, n0 = (idx & 15) << 3;   // 16 x 8-elem chunks per k-row
      union { uint4 q; ushort_t u[8]; } c;
      c.q = *(const uint4*)(Bp + (size_t)(k0 + kk) * ldb + bn0 + n0);
#pragma unroll
      for (int u = 0; u < 8; ++u) lB[(n0 + u) * 40 + kk] = c.u[u];
    }
    async_wait0();
    __syncthreads();

    union Frag { v16bf v; uint4 q[2]; };
    Frag fa[4], fb[2];
#pragma unroll
    for (int tm = 0; tm < 4; ++tm) {
      int row = wm * 64 + tm * 16 + laneLo;
      fa[tm].q[0] = *(const uint4*)&lA[row * 40 + 8 * laneHi];
      fa[tm].q[1] = *(const uint4*)&lA[row * 40 + 16 + 8 * laneHi];
    }
#pragma unroll
    for (int tn = 0; tn < 2; ++tn) {
      int col = wn * 32 + tn * 16 + laneLo;
      fb[tn].q[0] = *(const uint4*)&lB[col * 40 + 8 * laneHi];
      fb[tn].q[1] = *(const uint4*)&lB[col * 40 + 16 + 8 * laneHi];
    }
#pragma unroll
    for (int tm = 0; tm < 4; ++tm)
#pragma unroll
      for (int tn = 0; tn < 2; ++tn)
        acc[tm][tn] = __builtin_amdgcn_wmma_f32_16x16x32_bf16(
            false, fa[tm].v, false, fb[tn].v, (short)0, acc[tm][tn],
            false, false);
    __syncthreads();
  }

#pragma unroll
  for (int tm = 0; tm < 4; ++tm) {
#pragma unroll
    for (int tn = 0; tn < 2; ++tn) {
      int gcol = bn0 + wn * 32 + tn * 16 + laneLo;
#pragma unroll
      for (int r = 0; r < 8; ++r) {
        int grow = bm0 + wm * 64 + tm * 16 + r + 8 * laneHi;
        if (grow < M && gcol < N)
          C[(size_t)grow * ldc + gcol] = acc[tm][tn][r];
      }
    }
  }
}

// ---------------------------------------------------------------------------
// Row kernel: LN(gemm_out + bias_lin + bq + q_bias_raw[batch]) then L2-norm;
// writes bf16 (only consumed as GEMM operand).
// ---------------------------------------------------------------------------
__global__ __launch_bounds__(256) void add_ln_l2(
    const float* __restrict__ data, ushort_t* __restrict__ outb,
    const float* __restrict__ bias_lin, const float* __restrict__ bq,
    const float* __restrict__ qb, const float* __restrict__ g,
    const float* __restrict__ be, int rowsPerBatch)
{
  __shared__ float sm[256];
  const int row = blockIdx.x, t = threadIdx.x;
  const int b = row / rowsPerBatch;
  float v[4];
  float s = 0.f;
#pragma unroll
  for (int i = 0; i < 4; ++i) {
    int d = t + i * 256;
    v[i] = data[(size_t)row * D_ + d] + bias_lin[d] + bq[d] + qb[(size_t)b * D_ + d];
    s += v[i];
  }
  float mean = block_sum(s, sm) * (1.f / D_);
  float s2 = 0.f;
#pragma unroll
  for (int i = 0; i < 4; ++i) { float c = v[i] - mean; s2 += c * c; }
  float var = block_sum(s2, sm) * (1.f / D_);
  float rstd = rsqrtf(var + 1e-5f);
  float nrm = 0.f;
#pragma unroll
  for (int i = 0; i < 4; ++i) {
    int d = t + i * 256;
    v[i] = (v[i] - mean) * rstd * g[d] + be[d];
    nrm += v[i] * v[i];
  }
  float l2 = sqrtf(block_sum(nrm, sm));
  float sc = 1.f / fmaxf(l2, 1e-12f);
#pragma unroll
  for (int i = 0; i < 4; ++i)
    outb[(size_t)row * D_ + (t + i * 256)] = f2bf(v[i] * sc);
}

// Per-batch score stats -> low/high thresholds; rq = sum(vm)+sum(km)+1
__global__ __launch_bounds__(256) void stats_kernel(
    const float* __restrict__ scores, const unsigned char* __restrict__ vm,
    const unsigned char* __restrict__ km, float* __restrict__ low,
    float* __restrict__ high, float* __restrict__ rq)
{
  __shared__ float sm[256];
  const int b = blockIdx.x, t = threadIdx.x;
  const float* S = scores + (size_t)b * NV_ * NK_;
  float s = 0.f, ss = 0.f, cnt = 0.f;
  for (int it = 0; it < (NV_ * NK_) / 256; ++it) {
    int idx = t + it * 256;
    int i = idx >> 9, k = idx & (NK_ - 1);
    bool valid = vm[b * NV_ + i] && km[b * NK_ + k];
    float x = S[idx];
    if (valid) { s += x; ss += x * x; cnt += 1.f; }
  }
  float S1 = block_sum(s, sm);
  float S2 = block_sum(ss, sm);
  float C  = block_sum(cnt, sm);
  float svm = 0.f, skm = 0.f;
  for (int i = t; i < NV_; i += 256) svm += vm[b * NV_ + i] ? 1.f : 0.f;
  for (int k = t; k < NK_; k += 256) skm += km[b * NK_ + k] ? 1.f : 0.f;
  float SV = block_sum(svm, sm);
  float SK = block_sum(skm, sm);
  if (t == 0) {
    float c = fmaxf(C, 1.f);
    float mean = S1 / c;
    float var = fmaxf(S2 / c - mean * mean, 0.f);
    float sd = sqrtf(var);
    low[b]  = mean - 0.5f * sd;
    high[b] = mean + 0.5f * sd;
    rq[b]   = SV + SK + 1.f;
  }
}

// Masked top-8 + relevance-gated renormalized softmax (thread per source row)
__global__ __launch_bounds__(256) void topk_kernel(
    const float* __restrict__ scores, const unsigned char* __restrict__ tmask,
    const float* __restrict__ low, const float* __restrict__ high,
    float* __restrict__ w_out, int* __restrict__ idx_out,
    int nrows, int ntgt, int rs, int es)
{
  int gid = blockIdx.x * blockDim.x + threadIdx.x;
  if (gid >= B_ * nrows) return;
  int b = gid / nrows, r = gid - b * nrows;
  const float* base = scores + (size_t)b * NV_ * NK_ + (size_t)r * rs;
  const unsigned char* tm = tmask + (size_t)b * ntgt;

  float bv[TOPK_]; int bi[TOPK_];
#pragma unroll
  for (int j = 0; j < TOPK_; ++j) { bv[j] = -INFINITY; bi[j] = -1; }
  for (int j = 0; j < ntgt; ++j) {
    if (!tm[j]) continue;
    float v = base[(size_t)j * es];
    if (v > bv[TOPK_ - 1]) {
      int p = TOPK_ - 1;
      while (p > 0 && v > bv[p - 1]) { bv[p] = bv[p - 1]; bi[p] = bi[p - 1]; --p; }
      bv[p] = v; bi[p] = j;
    }
  }
  float lo = low[b], hi = high[b];
  float rel[TOPK_], e[TOPK_], esum = 0.f;
  float m = bv[0];
#pragma unroll
  for (int j = 0; j < TOPK_; ++j) {
    rel[j] = (bv[j] >= hi) ? 1.f : ((bv[j] >= lo) ? 0.5f : 0.f);
    e[j] = (rel[j] > 0.f) ? __expf(bv[j] - m) : 0.f;
    esum += e[j];
  }
  float inv = (esum > 0.f) ? (1.f / esum) : 0.f;
  float w[TOPK_], wsum = 0.f;
#pragma unroll
  for (int j = 0; j < TOPK_; ++j) { w[j] = e[j] * inv * rel[j]; wsum += w[j]; }
  float rn = 1.f / fmaxf(wsum, 1e-6f);
#pragma unroll
  for (int j = 0; j < TOPK_; ++j) {
    w_out[(size_t)gid * TOPK_ + j] = (bi[j] >= 0) ? w[j] * rn : 0.f;
    idx_out[(size_t)gid * TOPK_ + j] = (bi[j] >= 0) ? bi[j] : 0;
  }
}

// Scatter-max (weights >= 0, zeroed buffers -> int-bit atomicMax is order-safe)
static __device__ inline void atomicMaxPosF(float* addr, float v) {
  atomicMax((int*)addr, __float_as_int(v));
}
__global__ __launch_bounds__(256) void scatter_kernel(
    const float* __restrict__ w, const int* __restrict__ idx,
    const unsigned char* __restrict__ smask, float* __restrict__ cross,
    float* __restrict__ crossT, int nrows, int isVis)
{
  int gid = blockIdx.x * blockDim.x + threadIdx.x;
  if (gid >= B_ * nrows) return;
  int b = gid / nrows, r = gid - b * nrows;
  float sm = smask[(size_t)b * nrows + r] ? 1.f : 0.f;
  float* Cb  = cross  + (size_t)b * NV_ * NK_;
  float* CTb = crossT + (size_t)b * NK_ * NV_;
#pragma unroll
  for (int j = 0; j < TOPK_; ++j) {
    float ww = w[(size_t)gid * TOPK_ + j] * sm;
    if (ww <= 0.f) continue;
    int id = idx[(size_t)gid * TOPK_ + j];
    if (isVis) {
      atomicMaxPosF(&Cb[(size_t)r * NK_ + id], ww);
      atomicMaxPosF(&CTb[(size_t)id * NV_ + r], ww);
    } else {
      atomicMaxPosF(&Cb[(size_t)id * NK_ + r], ww);
      atomicMaxPosF(&CTb[(size_t)r * NV_ + id], ww);
    }
  }
}

// Adjacency row sums: out[row] = sum(mat[row,:]) + mask + 1 (self loop)
__global__ __launch_bounds__(256) void rowsum_kernel(
    const float* __restrict__ mat, const unsigned char* __restrict__ mask,
    float* __restrict__ out, int ncols, int rowsPerBatch)
{
  __shared__ float sm[256];
  int row = blockIdx.x, t = threadIdx.x;
  int b = row / rowsPerBatch, i = row - b * rowsPerBatch;
  float s = 0.f;
  for (int c = t; c < ncols; c += 256) s += mat[(size_t)row * ncols + c];
  float S = block_sum(s, sm);
  if (t == 0)
    out[row] = S + (mask[(size_t)b * rowsPerBatch + i] ? 1.f : 0.f) + 1.f;
}

// x0 = concat(visual, question, kg); writes fp32 x and bf16 x_b
__global__ __launch_bounds__(256) void build_x0(
    const float* __restrict__ vis, const float* __restrict__ q,
    const float* __restrict__ kg, float* __restrict__ x,
    ushort_t* __restrict__ xb)
{
  size_t gid = (size_t)blockIdx.x * 256 + threadIdx.x;
  size_t b = gid / ((size_t)NT_ * D_);
  size_t rem = gid - b * (size_t)NT_ * D_;
  int row = (int)(rem / D_), d = (int)(rem - (size_t)row * D_);
  float v;
  if (row < NV_)       v = vis[(b * NV_ + row) * D_ + d];
  else if (row == NV_) v = q[b * D_ + d];
  else                 v = kg[(b * NK_ + (row - KS_)) * D_ + d];
  x[gid] = v;
  xb[gid] = f2bf(v);
}

// y = normalized-adj @ x in factored form; y only feeds a GEMM -> bf16 out
__global__ __launch_bounds__(256) void combine_kernel(
    const float* __restrict__ P1, const float* __restrict__ P2,
    const float* __restrict__ x, const unsigned char* __restrict__ vm,
    const unsigned char* __restrict__ km, const float* __restrict__ r_vis,
    const float* __restrict__ r_kg, const float* __restrict__ rq,
    ushort_t* __restrict__ yb)
{
  int d = blockIdx.x * 256 + threadIdx.x;
  int row = blockIdx.y, b = blockIdx.z;
  const float* xb = x + (size_t)b * NT_ * D_;
  float xq = xb[(size_t)NV_ * D_ + d];
  float out;
  if (row < NV_) {
    float num = P1[((size_t)b * NV_ + row) * D_ + d]
              + (vm[b * NV_ + row] ? xq : 0.f)
              + xb[(size_t)row * D_ + d];
    out = num / fmaxf(r_vis[b * NV_ + row], 1e-6f);
  } else if (row == NV_) {
    float s = 0.f;
    for (int i = 0; i < NV_; ++i)
      if (vm[b * NV_ + i]) s += xb[(size_t)i * D_ + d];
    for (int k = 0; k < NK_; ++k)
      if (km[b * NK_ + k]) s += xb[(size_t)(KS_ + k) * D_ + d];
    out = (s + xq) / fmaxf(rq[b], 1e-6f);
  } else {
    int k = row - KS_;
    float num = P2[((size_t)b * NK_ + k) * D_ + d]
              + (km[b * NK_ + k] ? xq : 0.f)
              + xb[(size_t)row * D_ + d];
    out = num / fmaxf(r_kg[b * NK_ + k], 1e-6f);
  }
  yb[((size_t)b * NT_ + row) * D_ + d] = f2bf(out);
}

// x = LN(gelu(h + bg) + x); updates fp32 x in-place and bf16 x_b
__global__ __launch_bounds__(256) void gelu_res_ln(
    const float* __restrict__ h, const float* __restrict__ bg,
    float* __restrict__ x, ushort_t* __restrict__ xbf,
    const float* __restrict__ g, const float* __restrict__ be)
{
  __shared__ float sm[256];
  const int row = blockIdx.x, t = threadIdx.x;
  float v[4];
  float s = 0.f;
#pragma unroll
  for (int i = 0; i < 4; ++i) {
    int d = t + i * 256;
    float hv = h[(size_t)row * D_ + d] + bg[d];
    float ge = 0.5f * hv * (1.f + erff(hv * 0.70710678118654752f));
    v[i] = ge + x[(size_t)row * D_ + d];
    s += v[i];
  }
  float mean = block_sum(s, sm) * (1.f / D_);
  float s2 = 0.f;
#pragma unroll
  for (int i = 0; i < 4; ++i) { float c = v[i] - mean; s2 += c * c; }
  float var = block_sum(s2, sm) * (1.f / D_);
  float rstd = rsqrtf(var + 1e-5f);
#pragma unroll
  for (int i = 0; i < 4; ++i) {
    int d = t + i * 256;
    float o = (v[i] - mean) * rstd * g[d] + be[d];
    x[(size_t)row * D_ + d] = o;
    xbf[(size_t)row * D_ + d] = f2bf(o);
  }
}

// out = base + tanh(res) * (gout + bias)
__global__ __launch_bounds__(256) void out_epi(
    const float* __restrict__ base, const float* __restrict__ gout,
    const float* __restrict__ bias, const float* __restrict__ res,
    float* __restrict__ out, size_t total)
{
  size_t gid = (size_t)blockIdx.x * 256 + threadIdx.x;
  if (gid >= total) return;
  int d = (int)(gid & (D_ - 1));
  out[gid] = base[gid] + tanhf(res[0]) * (gout[gid] + bias[d]);
}

// ---------------------------------------------------------------------------
extern "C" void kernel_launch(void* const* d_in, const int* in_sizes, int n_in,
                              void* d_out, int out_size, void* d_ws, size_t ws_size,
                              hipStream_t stream) {
  (void)in_sizes; (void)n_in; (void)out_size; (void)ws_size;
  const float* visual = (const float*)d_in[0];
  const float* kg     = (const float*)d_in[1];
  const float* quest  = (const float*)d_in[2];
  const unsigned char* vm = (const unsigned char*)d_in[3];
  const unsigned char* km = (const unsigned char*)d_in[4];
  const float *Wq = (const float*)d_in[5],  *bq = (const float*)d_in[6];
  const float *Wv = (const float*)d_in[7],  *bv = (const float*)d_in[8];
  const float *Wk = (const float*)d_in[9],  *bk = (const float*)d_in[10];
  const float *Wg[2] = {(const float*)d_in[11], (const float*)d_in[13]};
  const float *bg[2] = {(const float*)d_in[12], (const float*)d_in[14]};
  const float *Wvo = (const float*)d_in[15], *bvo = (const float*)d_in[16];
  const float *Wko = (const float*)d_in[17], *bko = (const float*)d_in[18];
  const float *gvn = (const float*)d_in[19], *bvn = (const float*)d_in[20];
  const float *gkn = (const float*)d_in[21], *bkn = (const float*)d_in[22];
  const float *gn[2] = {(const float*)d_in[23], (const float*)d_in[25]};
  const float *bn[2] = {(const float*)d_in[24], (const float*)d_in[26]};
  const float *vres = (const float*)d_in[27], *kres = (const float*)d_in[28];

  // workspace carve-up with aliasing
  float* W = (float*)d_ws;
  size_t o = 0;
  auto alloc = [&](size_t nf) { float* p = W + o; o += (nf + 63) & ~(size_t)63; return p; };
  auto allocb = [&](size_t nb) { return (ushort_t*)alloc((nb + 1) / 2); };

  // fp32
  float* qb     = alloc((size_t)B_ * D_);
  float* linv   = alloc((size_t)B_ * NV_ * D_);   // gemm out v / P1 / Gv
  float* link   = alloc((size_t)B_ * NK_ * D_);   // gemm out k / P2 / Gk
  float* scores = alloc((size_t)B_ * NV_ * NK_);  // scores -> cross (alias)
  float* crossT = alloc((size_t)B_ * NK_ * NV_);
  float* w_vis  = alloc((size_t)B_ * NV_ * TOPK_);
  int*   i_vis  = (int*)alloc((size_t)B_ * NV_ * TOPK_);
  float* w_kg   = alloc((size_t)B_ * NK_ * TOPK_);
  int*   i_kg   = (int*)alloc((size_t)B_ * NK_ * TOPK_);
  float* lo     = alloc(B_);
  float* hi     = alloc(B_);
  float* rq     = alloc(B_);
  float* r_vis  = alloc((size_t)B_ * NV_);
  float* r_kg   = alloc((size_t)B_ * NK_);
  float* x      = alloc((size_t)B_ * NT_ * D_);
  float* h      = alloc((size_t)B_ * NT_ * D_);
  // bf16
  ushort_t* Wb[7];
  for (int i = 0; i < 7; ++i) Wb[i] = allocb((size_t)D_ * D_);
  ushort_t* quest_b = allocb((size_t)B_ * D_);
  ushort_t* vis_b   = allocb((size_t)B_ * NV_ * D_);   // -> vq_b (alias)
  ushort_t* kg_b    = allocb((size_t)B_ * NK_ * D_);   // -> kq_b (alias)
  ushort_t* cross_b  = allocb((size_t)B_ * NV_ * NK_);
  ushort_t* crossT_b = allocb((size_t)B_ * NK_ * NV_);
  ushort_t* x_b = allocb((size_t)B_ * NT_ * D_);
  ushort_t* y_b = allocb((size_t)B_ * NT_ * D_);

  float* cross = scores;
  float* P1 = linv, *P2 = link, *Gv = linv, *Gk = link;
  ushort_t* vq_b = vis_b;
  ushort_t* kq_b = kg_b;

  auto cvt = [&](const float* s, ushort_t* d, size_t n) {
    cvt_bf16<<<(int)((n / 4 + 255) / 256), 256, 0, stream>>>(s, d, n / 4);
  };

  // 0) one-time fp32 -> bf16 conversions of GEMM operands
  const float* Wsrc[7] = {Wq, Wv, Wk, Wg[0], Wg[1], Wvo, Wko};
  for (int i = 0; i < 7; ++i) cvt(Wsrc[i], Wb[i], (size_t)D_ * D_);
  cvt(quest,  quest_b, (size_t)B_ * D_);
  cvt(visual, vis_b,   (size_t)B_ * NV_ * D_);
  cvt(kg,     kg_b,    (size_t)B_ * NK_ * D_);

  // 1) q_bias raw = question @ Wq^T (bias folded into LN-prep)
  gemm_bf16_nt<<<dim3(D_ / 128, 1, 1), 256, 0, stream>>>(
      quest_b, 0, Wb[0], 0, qb, 0, B_, D_, D_, D_, D_, D_);

  // 2) vq = l2norm(LN(visual @ Wv^T + bv + bq + qb))   (bf16 out)
  gemm_bf16_nt<<<dim3(D_ / 128, (B_ * NV_) / 128, 1), 256, 0, stream>>>(
      vis_b, 0, Wb[1], 0, linv, 0, B_ * NV_, D_, D_, D_, D_, D_);
  add_ln_l2<<<B_ * NV_, 256, 0, stream>>>(linv, vq_b, bv, bq, qb, gvn, bvn, NV_);

  // 3) kq = l2norm(LN(kg @ Wk^T + bk + bq + qb))       (bf16 out)
  gemm_bf16_nt<<<dim3(D_ / 128, (B_ * NK_) / 128, 1), 256, 0, stream>>>(
      kg_b, 0, Wb[2], 0, link, 0, B_ * NK_, D_, D_, D_, D_, D_);
  add_ln_l2<<<B_ * NK_, 256, 0, stream>>>(link, kq_b, bk, bq, qb, gkn, bkn, NK_);

  // 4) scores[b] = vq[b] @ kq[b]^T (batched NT)
  gemm_bf16_nt<<<dim3(NK_ / 128, NV_ / 128, B_), 256, 0, stream>>>(
      vq_b, (size_t)NV_ * D_, kq_b, (size_t)NK_ * D_, scores,
      (size_t)NV_ * NK_, NV_, NK_, D_, D_, D_, NK_);

  // 5) stats, 6) top-8 both directions
  stats_kernel<<<B_, 256, 0, stream>>>(scores, vm, km, lo, hi, rq);
  topk_kernel<<<(B_ * NV_ + 255) / 256, 256, 0, stream>>>(
      scores, km, lo, hi, w_vis, i_vis, NV_, NK_, NK_, 1);
  topk_kernel<<<(B_ * NK_ + 255) / 256, 256, 0, stream>>>(
      scores, vm, lo, hi, w_kg, i_kg, NK_, NV_, 1, NK_);

  // 7) cross = max(vis, kg^T) * valid (scatter-max; scores now dead)
  hipMemsetAsync(cross,  0, (size_t)B_ * NV_ * NK_ * sizeof(float), stream);
  hipMemsetAsync(crossT, 0, (size_t)B_ * NK_ * NV_ * sizeof(float), stream);
  scatter_kernel<<<(B_ * NV_ + 255) / 256, 256, 0, stream>>>(
      w_vis, i_vis, vm, cross, crossT, NV_, 1);
  scatter_kernel<<<(B_ * NK_ + 255) / 256, 256, 0, stream>>>(
      w_kg, i_kg, km, cross, crossT, NK_, 0);

  // 8) adjacency row sums + bf16 copies of cross/crossT
  rowsum_kernel<<<B_ * NV_, 256, 0, stream>>>(cross, vm, r_vis, NK_, NV_);
  rowsum_kernel<<<B_ * NK_, 256, 0, stream>>>(crossT, km, r_kg, NV_, NK_);
  cvt(cross,  cross_b,  (size_t)B_ * NV_ * NK_);
  cvt(crossT, crossT_b, (size_t)B_ * NK_ * NV_);

  // 9) x0 = concat(visual, question, kg)
  build_x0<<<(int)(((size_t)B_ * NT_ * D_) / 256), 256, 0, stream>>>(
      visual, quest, kg, x, x_b);

  // 10) two GCN layers with factored adjacency
  for (int l = 0; l < 2; ++l) {
    gemm_bf16_nn<<<dim3(D_ / 128, NV_ / 128, B_), 256, 0, stream>>>(
        cross_b, (size_t)NV_ * NK_, x_b + (size_t)KS_ * D_, (size_t)NT_ * D_,
        P1, (size_t)NV_ * D_, NV_, D_, NK_, NK_, D_, D_);
    gemm_bf16_nn<<<dim3(D_ / 128, NK_ / 128, B_), 256, 0, stream>>>(
        crossT_b, (size_t)NK_ * NV_, x_b, (size_t)NT_ * D_,
        P2, (size_t)NK_ * D_, NK_, D_, NV_, NV_, D_, D_);
    combine_kernel<<<dim3(D_ / 256, NT_, B_), 256, 0, stream>>>(
        P1, P2, x, vm, km, r_vis, r_kg, rq, y_b);
    gemm_bf16_nt<<<dim3(D_ / 128, (B_ * NT_ + 127) / 128, 1), 256, 0, stream>>>(
        y_b, 0, Wb[3 + l], 0, h, 0, B_ * NT_, D_, D_, D_, D_, D_);
    gelu_res_ln<<<B_ * NT_, 256, 0, stream>>>(h, bg[l], x, x_b, gn[l], bn[l]);
  }

  // 11) output projections + gated residual
  float* v_out = (float*)d_out;
  float* k_out = (float*)d_out + (size_t)B_ * NV_ * D_;
  gemm_bf16_nt<<<dim3(D_ / 128, NV_ / 128, B_), 256, 0, stream>>>(
      x_b, (size_t)NT_ * D_, Wb[5], 0, Gv, (size_t)NV_ * D_, NV_, D_, D_, D_, D_, D_);
  out_epi<<<(int)(((size_t)B_ * NV_ * D_ + 255) / 256), 256, 0, stream>>>(
      visual, Gv, bvo, vres, v_out, (size_t)B_ * NV_ * D_);
  gemm_bf16_nt<<<dim3(D_ / 128, NK_ / 128, B_), 256, 0, stream>>>(
      x_b + (size_t)KS_ * D_, (size_t)NT_ * D_, Wb[6], 0, Gk, (size_t)NK_ * D_,
      NK_, D_, D_, D_, D_, D_);
  out_epi<<<(int)(((size_t)B_ * NK_ * D_ + 255) / 256), 256, 0, stream>>>(
      kg, Gk, bko, kres, k_out, (size_t)B_ * NK_ * D_);
}
